// WindowAttention_19344532701409
// MI455X (gfx1250) — matrix operand order repaired and checked by reading, not compiled, not run on hardware
//
#include <hip/hip_runtime.h>

// ---------------------------------------------------------------------------
// Window attention (Swin) for MI455X / gfx1250.
//   B_=4096 windows, N=49 tokens, C=512, H=16 heads, hd=32, nW=64 masks.
// Strategy: bf16 data + v_wmma_f32_16x16x32_bf16 (f32 accum), f32 softmax.
//   k1: qkv = x @ qkv_w + b         (f32->bf16 staged tiled GEMM, scatter q/k/v)
//   k2: per (window,head) wave: S = qK^T*scale+bias+mask, softmax, P@V
//   k3: out = attn_out @ proj_w + b (async-to-LDS A staging, f32 out)
// Grid axes: column-blocks on blockIdx.x (fastest) so blocks sharing an A-tile
// dispatch adjacently and A is served from the 192MB L2 instead of re-read
// from HBM (12x / 4x traffic reduction on the dominant GEMM input).
// ---------------------------------------------------------------------------

typedef __attribute__((ext_vector_type(16))) __bf16 v16bf;
typedef __attribute__((ext_vector_type(8)))  __bf16 v8bf;
typedef __attribute__((ext_vector_type(4)))  __bf16 v4bf;
typedef __attribute__((ext_vector_type(8)))  float  v8f;

#define WMMA_BF16(A, B, C) \
  __builtin_amdgcn_wmma_f32_16x16x32_bf16(false, (A), false, (B), (short)0, (C), false, false)

static __device__ __forceinline__ v16bf combine8(v8bf lo, v8bf hi) {
  v16bf r;
#pragma unroll
  for (int i = 0; i < 8; i++) { r[i] = lo[i]; r[i + 8] = hi[i]; }
  return r;
}

// ===========================================================================
// Kernel 1: qkv = x[200704x512] @ qkv_w[512x1536] + qkv_b, scattered to
// Q/K/V buffers laid out [B(4096)][H(16)][N(49)][hd(32)] in bf16.
// Block: 256 thr = 8 waves (2x4), tile 128x128, K-step 32, double-buffered LDS.
// ===========================================================================
__global__ __launch_bounds__(256) void qkv_gemm_kernel(
    const float* __restrict__ X, const float* __restrict__ W,
    const float* __restrict__ Bias,
    __bf16* __restrict__ Qo, __bf16* __restrict__ Ko, __bf16* __restrict__ Vo)
{
  __shared__ __attribute__((aligned(16))) __bf16 As[2][128][40];  // [row][k], pad 8
  __shared__ __attribute__((aligned(16))) __bf16 Bs[2][128][40];  // B^T: [col][k]
  const int tid  = threadIdx.x;
  const int lane = tid & 31, wid = tid >> 5;
  const int wm = wid >> 2, wn = wid & 3;          // 2 x 4 wave grid
  const int hl = lane >> 4, l15 = lane & 15;      // half-wave id, lane-in-half
  const unsigned row0 = blockIdx.y * 128;         // rows on Y (slow axis)
  const int      col0 = blockIdx.x * 128;         // cols on X: A-tile sharers adjacent

  auto stage = [&](int kt, int bi) {
    // A tile: 128x32 f32 -> bf16   (1024 float4 over 256 threads)
#pragma unroll
    for (int it = 0; it < 4; it++) {
      int f = tid + it * 256;
      int r = f >> 3;
      int c = (f & 7) * 4;
      float4 a4 = *(const float4*)(X + (size_t)(row0 + r) * 512 + kt * 32 + c);
      v4bf a;
      a[0] = (__bf16)a4.x; a[1] = (__bf16)a4.y; a[2] = (__bf16)a4.z; a[3] = (__bf16)a4.w;
      *(v4bf*)&As[bi][r][c] = a;
    }
    // B tile: 32x128 f32, transposed into Bs[col][k]
#pragma unroll
    for (int it = 0; it < 4; it++) {
      int f  = tid + it * 256;
      int kr = f >> 5;
      int c  = (f & 31) * 4;
      float4 w4 = *(const float4*)(W + (size_t)(kt * 32 + kr) * 1536 + col0 + c);
      Bs[bi][c + 0][kr] = (__bf16)w4.x;
      Bs[bi][c + 1][kr] = (__bf16)w4.y;
      Bs[bi][c + 2][kr] = (__bf16)w4.z;
      Bs[bi][c + 3][kr] = (__bf16)w4.w;
    }
  };

  v8f acc[4][2];
  v8f zf = {};
#pragma unroll
  for (int i = 0; i < 4; i++)
#pragma unroll
    for (int j = 0; j < 2; j++) acc[i][j] = zf;

  stage(0, 0);
  for (int kt = 0; kt < 16; kt++) {
    __syncthreads();
    if (kt + 1 < 16) stage(kt + 1, (kt + 1) & 1);
    const int bi = kt & 1;
    v16bf Af[4], Bf[2];
#pragma unroll
    for (int ti = 0; ti < 4; ti++) {
      int r  = wm * 64 + ti * 16 + l15;
      int k0 = hl * 8;                      // A layout: K runs [k0..k0+7],[k0+16..k0+23]
      v8bf lo = *(const v8bf*)&As[bi][r][k0];
      v8bf hi = *(const v8bf*)&As[bi][r][k0 + 16];
      Af[ti] = combine8(lo, hi);
    }
#pragma unroll
    for (int tj = 0; tj < 2; tj++) {
      int cc = wn * 32 + tj * 16 + l15;
      int k0 = hl * 8;                      // B layout: K runs [k0..k0+7],[k0+16..k0+23]
      v8bf lo = *(const v8bf*)&Bs[bi][cc][k0];
      v8bf hi = *(const v8bf*)&Bs[bi][cc][k0 + 16];
      Bf[tj] = combine8(lo, hi);
    }
#pragma unroll
    for (int ti = 0; ti < 4; ti++)
#pragma unroll
      for (int tj = 0; tj < 2; tj++)
        acc[ti][tj] = WMMA_BF16(Af[ti], Bf[tj], acc[ti][tj]);
  }

  // Epilogue: scatter into Q/K/V [b][h][n][d]
#pragma unroll
  for (int ti = 0; ti < 4; ti++)
#pragma unroll
    for (int tj = 0; tj < 2; tj++)
#pragma unroll
      for (int r = 0; r < 8; r++) {
        unsigned R = row0 + wm * 64 + ti * 16 + r + hl * 8;   // C layout: M = r + 8*hl
        int J = col0 + wn * 32 + tj * 16 + l15;               //           N = l15
        float val = acc[ti][tj][r] + Bias[J];
        int which = J >> 9;            // 0=q 1=k 2=v
        int c  = J & 511;
        int hh = c >> 5, d = c & 31;
        unsigned bwin = R / 49u;
        unsigned n    = R - bwin * 49u;
        size_t dst = ((size_t)(bwin * 16u + hh) * 49u + n) * 32u + d;
        __bf16 bv = (__bf16)val;
        if (which == 0)      Qo[dst] = bv;
        else if (which == 1) Ko[dst] = bv;
        else                 Vo[dst] = bv;
      }
}

// ===========================================================================
// Kernel 2: per-(window,head) attention. 1 wave per pair, 4 waves / block.
// S(64x64 padded) via 16 WMMAs (K=32 exactly = hd), f32 softmax with bias+mask,
// P through LDS -> A-fragments, P@V via 16 WMMAs. Output [B][N][C] bf16.
// ===========================================================================
__global__ __launch_bounds__(128) void attn_win_kernel(
    const __bf16* __restrict__ Q, const __bf16* __restrict__ K,
    const __bf16* __restrict__ V, const float* __restrict__ Mask,
    const float* __restrict__ BiasTab, __bf16* __restrict__ AO)
{
  __shared__ __attribute__((aligned(16))) __bf16 Plds[4][64][72];
  const int lane = threadIdx.x & 31;
  const int wid  = threadIdx.x >> 5;
  const int pair = blockIdx.x * 4 + wid;      // 0 .. 65535
  const int b = pair >> 4;
  const int h = pair & 15;
  const int hl = lane >> 4, l15 = lane & 15;
  const __bf16* q = Q + (size_t)pair * (49 * 32);
  const __bf16* k = K + (size_t)pair * (49 * 32);
  const __bf16* v = V + (size_t)pair * (49 * 32);

  // --- load A(q) and B(k^T) fragments directly from global ---
  v16bf Aq[4], Bk[4];
  {
    const int ak0 = hl * 8;
#pragma unroll
    for (int mi = 0; mi < 4; mi++) {
      int row = mi * 16 + l15; row = row > 48 ? 48 : row;
      v8bf lo = *(const v8bf*)(q + row * 32 + ak0);
      v8bf hi = *(const v8bf*)(q + row * 32 + ak0 + 16);
      Aq[mi] = combine8(lo, hi);
    }
    const int bk0 = hl * 8;
#pragma unroll
    for (int ni = 0; ni < 4; ni++) {
      int col = ni * 16 + l15; col = col > 48 ? 48 : col;   // key index
      v8bf lo = *(const v8bf*)(k + col * 32 + bk0);
      v8bf hi = *(const v8bf*)(k + col * 32 + bk0 + 16);
      Bk[ni] = combine8(lo, hi);                            // K-dim = head dim d
    }
  }

  // --- S = q @ k^T ---
  v8f S[4][4];
  v8f zf = {};
#pragma unroll
  for (int mi = 0; mi < 4; mi++)
#pragma unroll
    for (int ni = 0; ni < 4; ni++)
      S[mi][ni] = WMMA_BF16(Aq[mi], Bk[ni], zf);

  // --- scale + relative-position bias + window mask ---
  const float scale = 0.17677669529663687f;   // 32^-0.5
  const float* mrow = Mask + (size_t)(b & 63) * 49 * 49;
#pragma unroll
  for (int mi = 0; mi < 4; mi++)
#pragma unroll
    for (int ni = 0; ni < 4; ni++)
#pragma unroll
      for (int r = 0; r < 8; r++) {
        int M  = mi * 16 + r + hl * 8;
        int Nc = ni * 16 + l15;
        float val = S[mi][ni][r] * scale;
        if (M < 49 && Nc < 49) {
          int yn = M / 7,  xn = M - yn * 7;
          int ym = Nc / 7, xm = Nc - ym * 7;
          int idx = (yn - ym + 6) * 13 + (xn - xm + 6);
          val += BiasTab[idx * 16 + h] + mrow[M * 49 + Nc];
        } else {
          val = -1e30f;
        }
        S[mi][ni][r] = val;
      }

  // --- softmax over each row: row lives in one 16-lane half, 4 regs ---
#pragma unroll
  for (int mi = 0; mi < 4; mi++)
#pragma unroll
    for (int r = 0; r < 8; r++) {
      float mx = S[mi][0][r];
#pragma unroll
      for (int ni = 1; ni < 4; ni++) mx = fmaxf(mx, S[mi][ni][r]);
#pragma unroll
      for (int t = 1; t < 16; t <<= 1) mx = fmaxf(mx, __shfl_xor(mx, t, 16));
      float sum = 0.f;
#pragma unroll
      for (int ni = 0; ni < 4; ni++) {
        float e = __expf(S[mi][ni][r] - mx);
        S[mi][ni][r] = e;
        sum += e;
      }
#pragma unroll
      for (int t = 1; t < 16; t <<= 1) sum += __shfl_xor(sum, t, 16);
      float inv = 1.0f / sum;
#pragma unroll
      for (int ni = 0; ni < 4; ni++) S[mi][ni][r] *= inv;
    }

  // --- P -> LDS (bf16) so we can re-shape into A-fragment layout ---
  __bf16 (*pt)[72] = Plds[wid];
#pragma unroll
  for (int mi = 0; mi < 4; mi++)
#pragma unroll
    for (int ni = 0; ni < 4; ni++)
#pragma unroll
      for (int r = 0; r < 8; r++) {
        int M  = mi * 16 + r + hl * 8;
        int Nc = ni * 16 + l15;
        pt[M][Nc] = (__bf16)S[mi][ni][r];
      }
  // same-wave DS ops are in-order; loads below see the stores.

  // --- O = P(64x64) @ V(64x32): 2 K-steps of 32 ---
  v8f O[4][2];
#pragma unroll
  for (int i = 0; i < 4; i++) { O[i][0] = zf; O[i][1] = zf; }
#pragma unroll
  for (int ki = 0; ki < 2; ki++) {
    v16bf Ap[4];
#pragma unroll
    for (int mi = 0; mi < 4; mi++) {
      int row = mi * 16 + l15;
      int k0  = ki * 32 + hl * 16;
      v8bf lo = *(const v8bf*)&pt[row][k0];
      v8bf hi = *(const v8bf*)&pt[row][k0 + 8];
      Ap[mi] = combine8(lo, hi);
    }
    v16bf Bv[2];
#pragma unroll
    for (int ni = 0; ni < 2; ni++) {
      int d  = ni * 16 + l15;
      int m0 = ki * 32 + hl * 16;
#pragma unroll
      for (int i = 0; i < 16; i++) {
        int m = m0 + i; m = m > 48 ? 48 : m;   // padded rows multiply with P==0
        Bv[ni][i] = v[m * 32 + d];
      }
    }
#pragma unroll
    for (int mi = 0; mi < 4; mi++)
#pragma unroll
      for (int ni = 0; ni < 2; ni++)
        O[mi][ni] = WMMA_BF16(Ap[mi], Bv[ni], O[mi][ni]);
  }

  // --- store attn_out[b][n][h*32+d] ---
#pragma unroll
  for (int mi = 0; mi < 4; mi++)
#pragma unroll
    for (int ni = 0; ni < 2; ni++)
#pragma unroll
      for (int r = 0; r < 8; r++) {
        int M = mi * 16 + r + hl * 8;
        if (M < 49) {
          int d = ni * 16 + l15;
          AO[((size_t)b * 49 + M) * 512 + h * 32 + d] = (__bf16)O[mi][ni][r];
        }
      }
}

// ===========================================================================
// Kernel 3: out = attn_out[200704x512](bf16) @ proj_w[512x512] + proj_b (f32).
// A tiles staged with global_load_async_to_lds_b128 (ASYNCcnt path).
// ===========================================================================
__global__ __launch_bounds__(256) void proj_gemm_kernel(
    const __bf16* __restrict__ A, const float* __restrict__ W,
    const float* __restrict__ Bias, float* __restrict__ Out)
{
  __shared__ __attribute__((aligned(16))) __bf16 As[2][128][40];
  __shared__ __attribute__((aligned(16))) __bf16 Bs[2][128][40];  // B^T
  const int tid  = threadIdx.x;
  const int lane = tid & 31, wid = tid >> 5;
  const int wm = wid >> 2, wn = wid & 3;
  const int hl = lane >> 4, l15 = lane & 15;
  const unsigned row0 = blockIdx.y * 128;   // rows on Y (slow axis)
  const int      col0 = blockIdx.x * 128;   // cols on X: A-tile sharers adjacent

  auto stageA = [&](int kt, int bi) {
    // 128x32 bf16 tile = 512 x 16B chunks, async global->LDS, 2 per thread
#pragma unroll
    for (int it = 0; it < 2; it++) {
      int f = tid + it * 256;
      int r = f >> 2;
      int c = (f & 3) * 8;
      const __bf16* src = A + (size_t)(row0 + r) * 512 + kt * 32 + c;
      unsigned lofs = (unsigned)(size_t)&As[bi][r][c];   // low 32 bits = LDS offset
      asm volatile("global_load_async_to_lds_b128 %0, %1, off"
                   :: "v"(lofs), "v"(src) : "memory");
    }
  };
  auto stageB = [&](int kt, int bi) {
#pragma unroll
    for (int it = 0; it < 4; it++) {
      int f  = tid + it * 256;
      int kr = f >> 5;
      int c  = (f & 31) * 4;
      float4 w4 = *(const float4*)(W + (size_t)(kt * 32 + kr) * 512 + col0 + c);
      Bs[bi][c + 0][kr] = (__bf16)w4.x;
      Bs[bi][c + 1][kr] = (__bf16)w4.y;
      Bs[bi][c + 2][kr] = (__bf16)w4.z;
      Bs[bi][c + 3][kr] = (__bf16)w4.w;
    }
  };

  v8f acc[4][2];
  v8f zf = {};
#pragma unroll
  for (int i = 0; i < 4; i++)
#pragma unroll
    for (int j = 0; j < 2; j++) acc[i][j] = zf;

  stageA(0, 0);
  stageB(0, 0);
  for (int kt = 0; kt < 16; kt++) {
    asm volatile("s_wait_asynccnt 0" ::: "memory");  // my async LDS writes done
    __syncthreads();                                 // publish to all waves
    if (kt + 1 < 16) { stageA(kt + 1, (kt + 1) & 1); stageB(kt + 1, (kt + 1) & 1); }
    const int bi = kt & 1;
    v16bf Af[4], Bf[2];
#pragma unroll
    for (int ti = 0; ti < 4; ti++) {
      int r  = wm * 64 + ti * 16 + l15;
      int k0 = hl * 8;
      v8bf lo = *(const v8bf*)&As[bi][r][k0];
      v8bf hi = *(const v8bf*)&As[bi][r][k0 + 16];
      Af[ti] = combine8(lo, hi);
    }
#pragma unroll
    for (int tj = 0; tj < 2; tj++) {
      int cc = wn * 32 + tj * 16 + l15;
      int k0 = hl * 8;
      v8bf lo = *(const v8bf*)&Bs[bi][cc][k0];
      v8bf hi = *(const v8bf*)&Bs[bi][cc][k0 + 16];
      Bf[tj] = combine8(lo, hi);
    }
#pragma unroll
    for (int ti = 0; ti < 4; ti++)
#pragma unroll
      for (int tj = 0; tj < 2; tj++)
        acc[ti][tj] = WMMA_BF16(Af[ti], Bf[tj], acc[ti][tj]);
  }

#pragma unroll
  for (int ti = 0; ti < 4; ti++)
#pragma unroll
    for (int tj = 0; tj < 2; tj++)
#pragma unroll
      for (int r = 0; r < 8; r++) {
        unsigned R = row0 + wm * 64 + ti * 16 + r + hl * 8;
        int J = col0 + wn * 32 + tj * 16 + l15;
        Out[(size_t)R * 512 + J] = acc[ti][tj][r] + Bias[J];
      }
}

// ===========================================================================
extern "C" void kernel_launch(void* const* d_in, const int* in_sizes, int n_in,
                              void* d_out, int out_size, void* d_ws, size_t ws_size,
                              hipStream_t stream)
{
  const float* x          = (const float*)d_in[0];
  const float* mask       = (const float*)d_in[1];
  const float* qkv_w      = (const float*)d_in[2];
  const float* qkv_b      = (const float*)d_in[3];
  const float* proj_w     = (const float*)d_in[4];
  const float* proj_b     = (const float*)d_in[5];
  const float* bias_table = (const float*)d_in[6];

  const size_t E = (size_t)4096 * 16 * 49 * 32;  // 102,760,448 elems per buffer
  __bf16* Qb = (__bf16*)d_ws;
  __bf16* Kb = Qb + E;
  __bf16* Vb = Kb + E;
  __bf16* Ao = Vb + E;   // [200704 x 512] bf16 (same element count)
  // total workspace use: 4*E*2 = 822,083,584 bytes

  // cols on grid X (fastest) -> A-tile reuse through L2
  qkv_gemm_kernel<<<dim3(12, 1568), 256, 0, stream>>>(x, qkv_w, qkv_b, Qb, Kb, Vb);
  attn_win_kernel<<<dim3(16384), 128, 0, stream>>>(Qb, Kb, Vb, mask, bias_table, Ao);
  proj_gemm_kernel<<<dim3(4, 1568), 256, 0, stream>>>(Ao, proj_w, proj_b, (float*)d_out);
}
